// TemplatePointwiseAttention_57561151701401
// MI455X (gfx1250) — matrix operand order repaired and checked by reading, not compiled, not run on hardware
//
#include <hip/hip_runtime.h>
#include <hip/hip_bf16.h>

typedef __attribute__((ext_vector_type(16))) __bf16 v16bf;
typedef __attribute__((ext_vector_type(8)))  __bf16 v8bf;
typedef __attribute__((ext_vector_type(4)))  __bf16 v4bf;
typedef __attribute__((ext_vector_type(2)))  __bf16 v2bf;
typedef __attribute__((ext_vector_type(8)))  float  v8f;
typedef __attribute__((ext_vector_type(4)))  float  v4f;
typedef __attribute__((ext_vector_type(2)))  float  v2f;

#define N_RES    512
#define N_PAIRS  (512 * 512)
#define N_TEMPL  4
#define INFB     10000.0f

// ---- packed bf16 conversion helpers (native cvt, RNE) ----
__device__ __forceinline__ unsigned long long pack4bf(v4f f) {
    return __builtin_bit_cast(unsigned long long, __builtin_convertvector(f, v4bf));
}
__device__ __forceinline__ unsigned pack2bf(float a, float b) {
    v2f f = {a, b};
    return __builtin_bit_cast(unsigned, __builtin_convertvector(f, v2bf));
}

// ---- WMMA wrapper ----
__device__ __forceinline__ v8f wmma_bf16(v16bf a, v16bf b, v8f c) {
    return __builtin_amdgcn_wmma_f32_16x16x32_bf16(
        /*neg_a=*/false, a, /*neg_b=*/false, b,
        /*c_mod=*/(short)0, c, /*reuse_a=*/false, /*reuse_b=*/false);
}

// A fragment (16x32 bf16, MxK) from LDS staging, row-major with `stride` halves.
// lane L: row = L&15; halves 0..7 = K kb0..kb0+7 ; halves 8..15 = K kb0+16..kb0+23
// where kb0 = kk*32 + (L<16 ? 0 : 8)   (per ISA 16-bit A layout)
__device__ __forceinline__ v16bf load_afrag(const unsigned short* buf, int stride,
                                            int kk, int lane) {
    int r  = lane & 15;
    int kb = kk * 32 + ((lane >> 4) << 3);
    const unsigned short* p = buf + r * stride + kb;
    v8bf lo = *(const v8bf*)(p);
    v8bf hi = *(const v8bf*)(p + 16);
    return __builtin_shufflevector(lo, hi, 0,1,2,3,4,5,6,7,8,9,10,11,12,13,14,15);
}

// B fragment (32x16 bf16, KxN) pre-swizzled in global scratch: contiguous 16 halves.
__device__ __forceinline__ v16bf load_bfrag(const unsigned short* base, int frag, int lane) {
    return *(const v16bf*)(base + ((size_t)frag * 32 + lane) * 16);
}

// D fragment (f32) -> bf16 row-major LDS tile
__device__ __forceinline__ void store_dfrag(unsigned short* base, int stride, int coloff,
                                            int mbase, int lane15, v8f acc) {
    v8bf d = __builtin_convertvector(acc, v8bf);
#pragma unroll
    for (int i = 0; i < 8; ++i)
        base[(mbase + i) * stride + coloff + lane15] =
            __builtin_bit_cast(unsigned short, (__bf16)d[i]);
}

// 16 contiguous bf16 from LDS -> 16 floats
__device__ __forceinline__ void load_row16(const unsigned short* p, float* f) {
    v8bf a = *(const v8bf*)p;
    v8bf b = *(const v8bf*)(p + 8);
    v8f fa = __builtin_convertvector(a, v8f);
    v8f fb = __builtin_convertvector(b, v8f);
#pragma unroll
    for (int i = 0; i < 8; ++i) { f[i] = fa[i]; f[8 + i] = fb[i]; }
}

// ---------------------------------------------------------------------------
// Kernel 1: pre-swizzle f32 weights into bf16 B-fragment layout in d_ws.
// Layout (halves): wq @ 0 (512 slots), wk @ 8192 (256), wv @ 12288 (256),
// wo @ 16384 (512). Slot = frag*32 + lane, 16 halves each.
// B-frag element h: K = kk*32 + (lane<16?0:8) + h + (h<8?0:8), N = j*16 + (lane&15).
// ---------------------------------------------------------------------------
__global__ void prep_weights_kernel(const float* __restrict__ wq,
                                    const float* __restrict__ wk,
                                    const float* __restrict__ wv,
                                    const float* __restrict__ wo,
                                    unsigned short* __restrict__ ws) {
    int id = blockIdx.x * blockDim.x + threadIdx.x;   // 0..1535
    const float* src;
    unsigned short* dst;
    int slot, ncols, ksteps;
    if (id < 512)       { src = wq; dst = ws;          slot = id;        ncols = 64;  ksteps = 4; }
    else if (id < 768)  { src = wk; dst = ws + 8192;   slot = id - 512;  ncols = 64;  ksteps = 2; }
    else if (id < 1024) { src = wv; dst = ws + 12288;  slot = id - 768;  ncols = 64;  ksteps = 2; }
    else                { src = wo; dst = ws + 16384;  slot = id - 1024; ncols = 128; ksteps = 2; }

    int lane = slot & 31;
    int frag = slot >> 5;
    int j  = frag / ksteps;
    int kk = frag % ksteps;
    int kb = kk * 32 + ((lane >> 4) << 3);     // +8 for upper half-wave
    int n  = j * 16 + (lane & 15);
    unsigned short* o = dst + (size_t)slot * 16;
#pragma unroll
    for (int h = 0; h < 16; ++h)
        o[h] = __builtin_bit_cast(unsigned short, (__bf16)src[(size_t)(kb + h + ((h >> 3) << 3)) * ncols + n]);
}

// ---------------------------------------------------------------------------
// Kernel 2: fused attention. 2 waves / block, each wave owns 16 pairs.
// ---------------------------------------------------------------------------
#define STAGE_STRIDE 136               // halves, 16B-aligned rows, padded
#define QKV_STRIDE   72                // halves, 16B-aligned rows, padded
#define STAGE_SZ     (16 * STAGE_STRIDE)   // 2176
#define QKV_SZ       (16 * QKV_STRIDE)     // 1152
#define WAVE_LDS     (STAGE_SZ + QKV_SZ + QKV_SZ + 4 * QKV_SZ)  // 9088 halves

__global__ __launch_bounds__(64) void
tpa_kernel(const float* __restrict__ t, const float* __restrict__ z,
           const float* __restrict__ mask, const float* __restrict__ bo,
           const unsigned short* __restrict__ wsq, const unsigned short* __restrict__ wsk,
           const unsigned short* __restrict__ wsv, const unsigned short* __restrict__ wso,
           float* __restrict__ out) {
    __shared__ __align__(16) unsigned short lds[2 * WAVE_LDS];

    const int wave   = threadIdx.x >> 5;
    const int lane   = threadIdx.x & 31;
    const int lane15 = lane & 15;
    const int mbase  = (lane >> 4) * 8;   // D-frag rows held by this half-wave
    const int hp     = (lane >> 4) * 2;   // heads handled in per-lane phase

    unsigned short* stage = lds + wave * WAVE_LDS;       // z / t / O staging (bf16)
    unsigned short* qs    = stage + STAGE_SZ;            // Q  [16][72]
    unsigned short* ks    = qs + QKV_SZ;                 // K  [16][72] (per template, reused)
    unsigned short* vs    = ks + QKV_SZ;                 // V  [4][16][72]

    const size_t pBase = ((size_t)blockIdx.x * 2 + wave) * 16;

    // ---- bias from template mask ----
    float bias[N_TEMPL];
#pragma unroll
    for (int s = 0; s < N_TEMPL; ++s) bias[s] = INFB * (mask[s] - 1.0f);

    // ---- stage z tile [16 x 128] -> bf16 LDS ----
#pragma unroll
    for (int i = 0; i < 16; ++i) {
        int idx4 = (lane + i * 32) * 4;      // 0..2044
        int r = idx4 >> 7, c = idx4 & 127;
        v4f v = *(const v4f*)(z + (pBase + r) * 128 + c);
        *(unsigned long long*)&stage[r * STAGE_STRIDE + c] = pack4bf(v);
    }
    asm volatile("s_wait_dscnt 0" ::: "memory");

    // ---- Q = Z @ wq  (K=128, N=64) ----
#pragma unroll
    for (int j = 0; j < 4; ++j) {
        v8f acc = {};
#pragma unroll
        for (int kk = 0; kk < 4; ++kk) {
            v16bf a = load_afrag(stage, STAGE_STRIDE, kk, lane);
            v16bf b = load_bfrag(wsq, j * 4 + kk, lane);
            acc = wmma_bf16(a, b, acc);
        }
        store_dfrag(qs, QKV_STRIDE, j * 16, mbase, lane15, acc);
    }
    asm volatile("s_wait_dscnt 0" ::: "memory");

    // ---- hoist this pair's q rows (scale folded in) ----
    float qrow[2][16];
#pragma unroll
    for (int th = 0; th < 2; ++th) {
        load_row16(qs + lane15 * QKV_STRIDE + (hp + th) * 16, qrow[th]);
#pragma unroll
        for (int c = 0; c < 16; ++c) qrow[th][c] *= 0.25f;   // 1/sqrt(16)
    }

    float sc[2][N_TEMPL];

    // ---- per-template: stage t, K-proj, scores, V-proj ----
    for (int s = 0; s < N_TEMPL; ++s) {
        const float* tp = t + ((size_t)s * N_PAIRS + pBase) * 64;
#pragma unroll
        for (int i = 0; i < 8; ++i) {
            int idx4 = (lane + i * 32) * 4;   // 0..1020
            int r = idx4 >> 6, c = idx4 & 63;
            v4f v = *(const v4f*)(tp + r * 64 + c);
            *(unsigned long long*)&stage[r * STAGE_STRIDE + c] = pack4bf(v);
        }
        if (s < N_TEMPL - 1)  // prefetch next template tile (global_prefetch_b8)
            __builtin_prefetch(tp + (size_t)N_PAIRS * 64 + (size_t)lane * 32, 0, 0);
        asm volatile("s_wait_dscnt 0" ::: "memory");

        // K-proj (K=64, N=64)
#pragma unroll
        for (int j = 0; j < 4; ++j) {
            v8f acc = {};
#pragma unroll
            for (int kk = 0; kk < 2; ++kk) {
                v16bf a = load_afrag(stage, STAGE_STRIDE, kk, lane);
                v16bf b = load_bfrag(wsk, j * 2 + kk, lane);
                acc = wmma_bf16(a, b, acc);
            }
            store_dfrag(ks, QKV_STRIDE, j * 16, mbase, lane15, acc);
        }
        asm volatile("s_wait_dscnt 0" ::: "memory");

        // scores: pair m = lane15, heads hp, hp+1
#pragma unroll
        for (int th = 0; th < 2; ++th) {
            float kr[16];
            load_row16(ks + lane15 * QKV_STRIDE + (hp + th) * 16, kr);
            float dot = 0.f;
#pragma unroll
            for (int c = 0; c < 16; ++c) dot += qrow[th][c] * kr[c];
            sc[th][s] = dot + bias[s];
        }

        // V-proj (K=64, N=64)
#pragma unroll
        for (int j = 0; j < 4; ++j) {
            v8f acc = {};
#pragma unroll
            for (int kk = 0; kk < 2; ++kk) {
                v16bf a = load_afrag(stage, STAGE_STRIDE, kk, lane);
                v16bf b = load_bfrag(wsv, j * 2 + kk, lane);
                acc = wmma_bf16(a, b, acc);
            }
            store_dfrag(vs + s * QKV_SZ, QKV_STRIDE, j * 16, mbase, lane15, acc);
        }
    }
    asm volatile("s_wait_dscnt 0" ::: "memory");

    // ---- softmax over S, weighted sum of V, write O (bf16) to stage ----
#pragma unroll
    for (int th = 0; th < 2; ++th) {
        int h = hp + th;
        float mx = sc[th][0];
#pragma unroll
        for (int s = 1; s < N_TEMPL; ++s) mx = fmaxf(mx, sc[th][s]);
        float e[N_TEMPL], sum = 0.f;
#pragma unroll
        for (int s = 0; s < N_TEMPL; ++s) { e[s] = __expf(sc[th][s] - mx); sum += e[s]; }
        float inv = 1.0f / sum;
#pragma unroll
        for (int s = 0; s < N_TEMPL; ++s) e[s] *= inv;

        float o[16];
#pragma unroll
        for (int c = 0; c < 16; ++c) o[c] = 0.f;
#pragma unroll
        for (int s = 0; s < N_TEMPL; ++s) {
            float vr[16];
            load_row16(vs + s * QKV_SZ + lane15 * QKV_STRIDE + h * 16, vr);
#pragma unroll
            for (int c = 0; c < 16; ++c) o[c] += e[s] * vr[c];
        }
#pragma unroll
        for (int c = 0; c < 8; ++c)
            *(unsigned*)&stage[lane15 * STAGE_STRIDE + h * 16 + 2 * c] =
                pack2bf(o[2 * c], o[2 * c + 1]);
    }
    asm volatile("s_wait_dscnt 0" ::: "memory");

    // ---- out = O @ wo + bo  (K=64, N=128), D stored straight to global ----
#pragma unroll
    for (int j = 0; j < 8; ++j) {
        float bj = bo[j * 16 + lane15];
        v8f acc = {bj, bj, bj, bj, bj, bj, bj, bj};
#pragma unroll
        for (int kk = 0; kk < 2; ++kk) {
            v16bf a = load_afrag(stage, STAGE_STRIDE, kk, lane);
            v16bf b = load_bfrag(wso, j * 2 + kk, lane);
            acc = wmma_bf16(a, b, acc);
        }
#pragma unroll
        for (int i = 0; i < 8; ++i)
            out[(pBase + mbase + i) * 128 + j * 16 + lane15] = acc[i];
    }
}

// ---------------------------------------------------------------------------
extern "C" void kernel_launch(void* const* d_in, const int* in_sizes, int n_in,
                              void* d_out, int out_size, void* d_ws, size_t ws_size,
                              hipStream_t stream) {
    const float* t    = (const float*)d_in[0];
    const float* z    = (const float*)d_in[1];
    const float* mask = (const float*)d_in[2];
    const float* wq   = (const float*)d_in[3];
    const float* wk   = (const float*)d_in[4];
    const float* wv   = (const float*)d_in[5];
    const float* wo   = (const float*)d_in[6];
    const float* bo   = (const float*)d_in[7];
    unsigned short* ws = (unsigned short*)d_ws;   // 24576 halves = 48KB used

    prep_weights_kernel<<<6, 256, 0, stream>>>(wq, wk, wv, wo, ws);

    // 262144 pairs / (2 waves * 16 pairs) = 8192 blocks of 64 threads
    tpa_kernel<<<8192, 64, 0, stream>>>(t, z, mask, bo,
                                        ws, ws + 8192, ws + 12288, ws + 16384,
                                        (float*)d_out);
}
